// GraphLSTMCell_13228499271721
// MI455X (gfx1250) — compile-verified, matrix operand order checked by reading, not run on hardware
//
#include <hip/hip_runtime.h>
#include <hip/hip_bf16.h>
#include <math.h>

// ---------- types for WMMA ----------
typedef __bf16  v16bf __attribute__((ext_vector_type(16)));
typedef __bf16  v8bf  __attribute__((ext_vector_type(8)));
typedef float   v8f   __attribute__((ext_vector_type(8)));

union BF16x16 { v16bf v; v8bf h[2]; };

#define IN_FEATS  128
#define OUT_FEATS 128
#define KDIM      512   // 4 concatenated 128-wide inputs
#define B_LDS_BYTES (OUT_FEATS * KDIM * 2)   // 128 KB bf16 weight tile per gate

__device__ __forceinline__ __bf16 f2bf(float f) {
    unsigned int u = __float_as_uint(f);
    u += 0x7FFFu + ((u >> 16) & 1u);          // round-to-nearest-even
    unsigned short s = (unsigned short)(u >> 16);
    return __builtin_bit_cast(__bf16, s);
}

__device__ __forceinline__ float sigmoidf(float x) {
    return 1.0f / (1.0f + expf(-x));
}

// ---------- 1. zero scratch (sums + degree) ----------
__global__ void zero_kernel(float* __restrict__ p, long long n) {
    long long i = (long long)blockIdx.x * blockDim.x + threadIdx.x;
    if (i < n) p[i] = 0.0f;
}

// ---------- 2. pack weights: Wcat[g][outCol][k] bf16, k-major (column-major W) ----------
__global__ void pack_w_kernel(const float* __restrict__ Wxs, const float* __restrict__ Wxn,
                              const float* __restrict__ Whs, const float* __restrict__ Whn,
                              __bf16* __restrict__ Wcat) {
    int idx = blockIdx.x * blockDim.x + threadIdx.x;   // [0, 4*128*512)
    if (idx >= 4 * OUT_FEATS * KDIM) return;
    int k = idx & (KDIM - 1);
    int o = (idx >> 9) & (OUT_FEATS - 1);
    int g = idx >> 16;
    float w;
    if      (k < 128) w = Wxs[((g * 128) + k      ) * 128 + o];
    else if (k < 256) w = Wxn[((g * 128) + k - 128) * 128 + o];
    else if (k < 384) w = Whs[((g * 128) + k - 256) * 128 + o];
    else              w = Whn[((g * 128) + k - 384) * 128 + o];
    Wcat[((size_t)(g * OUT_FEATS) + o) * KDIM + k] = f2bf(w);
}

// ---------- 3. edge scatter-add (one wave per edge) ----------
__global__ void edge_agg_kernel(const float* __restrict__ x, const float* __restrict__ h,
                                const int* __restrict__ src, const int* __restrict__ dst,
                                float* __restrict__ sum_x, float* __restrict__ sum_h,
                                float* __restrict__ deg, int nEdges) {
    int wave = (blockIdx.x * blockDim.x + threadIdx.x) >> 5;
    int lane = threadIdx.x & 31;
    if (wave >= nEdges) return;
    int s = src[wave];
    int d = dst[wave];
    const float* xs = x + (size_t)s * IN_FEATS;
    const float* hs = h + (size_t)s * OUT_FEATS;
    float* sx = sum_x + (size_t)d * IN_FEATS;
    float* sh = sum_h + (size_t)d * OUT_FEATS;
#pragma unroll
    for (int f = 0; f < 4; ++f) {
        int j = lane + f * 32;
        atomicAdd(&sx[j], xs[j]);
        atomicAdd(&sh[j], hs[j]);
    }
    if (lane == 0) atomicAdd(&deg[d], 1.0f);
}

// ---------- 4. build bf16 A = [x | mean(x) | h | mean(h)]  -> [N][512] ----------
__global__ void build_a_kernel(const float* __restrict__ x, const float* __restrict__ h,
                               const float* __restrict__ sum_x, const float* __restrict__ sum_h,
                               const float* __restrict__ deg, __bf16* __restrict__ A, int nNodes) {
    long long idx = (long long)blockIdx.x * blockDim.x + threadIdx.x;
    if (idx >= (long long)nNodes * 128) return;
    int n = (int)(idx >> 7);
    int f = (int)(idx & 127);
    float inv = 1.0f / fmaxf(deg[n], 1.0f);
    __bf16* a = A + (size_t)n * KDIM;
    size_t off = (size_t)n * 128 + f;
    a[f]       = f2bf(x[off]);
    a[128 + f] = f2bf(sum_x[off] * inv);
    a[256 + f] = f2bf(h[off]);
    a[384 + f] = f2bf(sum_h[off] * inv);
}

// ---------- 5. WMMA GEMM: S[g][N][128] = A[N][512] * Wcat[g] ----------
// Block = 8 waves sharing the gate's 128KB weight tile in LDS.
// Wave w: M-tile = blockIdx.x*4 + (w&3), column half = (w>>2)*64.
// Each wave: 4 accumulators (16x16), explicit 2-deep software pipeline on
// A/B fragments so next-step ds/global loads overlap current-step WMMAs.
__global__ void gemm_kernel(const __bf16* __restrict__ A, const __bf16* __restrict__ Wcat,
                            float* __restrict__ S, int nNodes) {
    extern __shared__ __bf16 Bs[];   // OUT_FEATS * KDIM elements (128 KB)
    int g = blockIdx.y;

    // cooperative stage: global -> LDS, 16B chunks
    {
        const uint4* srcp = (const uint4*)(Wcat + (size_t)g * OUT_FEATS * KDIM);
        uint4* dstp = (uint4*)Bs;
        const int chunks = B_LDS_BYTES / 16;          // 8192
#pragma unroll 4
        for (int i = threadIdx.x; i < chunks; i += 256)
            dstp[i] = srcp[i];
    }
    __syncthreads();

    int waveInBlock = threadIdx.x >> 5;
    int lane = threadIdx.x & 31;
    int mTile   = blockIdx.x * 4 + (waveInBlock & 3);  // uniform per wave
    int colHalf = waveInBlock >> 2;                    // 0 or 1 (64-col halves)
    if (mTile * 16 >= nNodes) return;

    int mBase = mTile * 16;
    int half  = lane >> 4;      // K-half select per ISA 16-bit A/B layout
    int lrow  = lane & 15;      // M row (A) / N col (B)

    const __bf16* Ap = A  + (size_t)(mBase + lrow) * KDIM + half * 8;
    const __bf16* Bb = Bs + (size_t)(colHalf * 64 + lrow) * KDIM + half * 8;

    v8f acc[4];
#pragma unroll
    for (int nt = 0; nt < 4; ++nt) acc[nt] = (v8f)0.0f;

    BF16x16 a[2];
    BF16x16 b[2][4];

    // prologue: stage k0 = 0
    a[0].h[0] = *(const v8bf*)(Ap);
    a[0].h[1] = *(const v8bf*)(Ap + 16);
#pragma unroll
    for (int nt = 0; nt < 4; ++nt) {
        const __bf16* Bp = Bb + nt * 16 * KDIM;
        b[0][nt].h[0] = *(const v8bf*)(Bp);
        b[0][nt].h[1] = *(const v8bf*)(Bp + 16);
    }

#pragma unroll
    for (int ks = 0; ks < 16; ++ks) {
        const int cur = ks & 1;
        const int nxt = cur ^ 1;
        if (ks < 15) {
            int k1 = (ks + 1) * 32;
            a[nxt].h[0] = *(const v8bf*)(Ap + k1);
            a[nxt].h[1] = *(const v8bf*)(Ap + k1 + 16);
#pragma unroll
            for (int nt = 0; nt < 4; ++nt) {
                const __bf16* Bp = Bb + nt * 16 * KDIM + k1;
                b[nxt][nt].h[0] = *(const v8bf*)(Bp);
                b[nxt][nt].h[1] = *(const v8bf*)(Bp + 16);
            }
        }
#pragma unroll
        for (int nt = 0; nt < 4; ++nt) {
            acc[nt] = __builtin_amdgcn_wmma_f32_16x16x32_bf16(
                false, a[cur].v, false, b[cur][nt].v, (short)0, acc[nt], false, false);
        }
    }

    // D layout: VGPR r -> M = r (lanes 0-15) / r+8 (lanes 16-31); N = lane&15
    float* Sg = S + ((size_t)g * nNodes + mBase) * OUT_FEATS + colHalf * 64;
#pragma unroll
    for (int nt = 0; nt < 4; ++nt) {
#pragma unroll
        for (int r = 0; r < 8; ++r) {
            Sg[(size_t)(r + half * 8) * OUT_FEATS + nt * 16 + lrow] = acc[nt][r];
        }
    }
}

// ---------- 6. fused LSTM gate elementwise ----------
__global__ void lstm_kernel(const float* __restrict__ S, const float* __restrict__ c,
                            const float* __restrict__ w_ci, const float* __restrict__ w_cf,
                            const float* __restrict__ w_co, const float* __restrict__ b_i,
                            const float* __restrict__ b_f, const float* __restrict__ b_c,
                            const float* __restrict__ b_o, float* __restrict__ out, int nNodes) {
    long long idx = (long long)blockIdx.x * blockDim.x + threadIdx.x;
    long long total = (long long)nNodes * OUT_FEATS;
    if (idx >= total) return;
    int j = (int)(idx & (OUT_FEATS - 1));
    float s0 = S[idx];
    float s1 = S[total + idx];
    float s2 = S[2 * total + idx];
    float s3 = S[3 * total + idx];
    float cv = c[idx];
    float ig = sigmoidf(s0 + w_ci[j] * cv + b_i[j]);
    float fg = sigmoidf(s1 + w_cf[j] * cv + b_f[j]);
    float ct = tanhf(s2 + b_c[j]);
    float nc = fg * cv + ig * ct;
    float og = sigmoidf(s3 + w_co[j] * nc + b_o[j]);
    float nh = og * tanhf(nc);
    out[idx]         = nh;   // new_h
    out[total + idx] = nc;   // new_c
}

extern "C" void kernel_launch(void* const* d_in, const int* in_sizes, int n_in,
                              void* d_out, int out_size, void* d_ws, size_t ws_size,
                              hipStream_t stream) {
    const float* x   = (const float*)d_in[0];
    const float* h   = (const float*)d_in[1];
    const float* c   = (const float*)d_in[2];
    const float* Wxs = (const float*)d_in[3];
    const float* Wxn = (const float*)d_in[4];
    const float* Whs = (const float*)d_in[5];
    const float* Whn = (const float*)d_in[6];
    const float* w_ci = (const float*)d_in[7];
    const float* w_cf = (const float*)d_in[8];
    const float* w_co = (const float*)d_in[9];
    const float* b_i  = (const float*)d_in[10];
    const float* b_f  = (const float*)d_in[11];
    const float* b_c  = (const float*)d_in[12];
    const float* b_o  = (const float*)d_in[13];
    const int* src = (const int*)d_in[14];
    const int* dst = (const int*)d_in[15];
    float* out = (float*)d_out;

    const int N = in_sizes[0] / IN_FEATS;   // 50000
    const int E = in_sizes[14];             // 800000

    // ---- workspace carve (256B aligned regions) ----
    char* ws = (char*)d_ws;
    size_t off = 0;
    auto carve = [&](size_t bytes) -> char* {
        char* p = ws + off;
        off += (bytes + 255) & ~(size_t)255;
        return p;
    };
    float*  sum_x = (float*)carve((size_t)N * IN_FEATS  * sizeof(float));
    float*  sum_h = (float*)carve((size_t)N * OUT_FEATS * sizeof(float));
    float*  deg   = (float*)carve((size_t)N * sizeof(float));
    float*  S     = (float*)carve((size_t)4 * N * OUT_FEATS * sizeof(float));
    __bf16* Abf   = (__bf16*)carve((size_t)N * KDIM * sizeof(__bf16));
    __bf16* Wcat  = (__bf16*)carve((size_t)4 * OUT_FEATS * KDIM * sizeof(__bf16));
    (void)ws_size;

    // 1. zero sums + degree
    {
        long long n1 = (long long)N * IN_FEATS;
        zero_kernel<<<dim3((unsigned)((n1 + 255) / 256)), dim3(256), 0, stream>>>(sum_x, n1);
        zero_kernel<<<dim3((unsigned)((n1 + 255) / 256)), dim3(256), 0, stream>>>(sum_h, n1);
        zero_kernel<<<dim3((unsigned)((N + 255) / 256)), dim3(256), 0, stream>>>(deg, N);
    }

    // 2. pack weights to bf16, column-major over K
    {
        int tot = 4 * OUT_FEATS * KDIM;
        pack_w_kernel<<<dim3((tot + 255) / 256), dim3(256), 0, stream>>>(Wxs, Wxn, Whs, Whn, Wcat);
    }

    // 3. edge scatter-add: one wave (32 lanes) per edge
    {
        int blocks = (E * 32 + 255) / 256;
        edge_agg_kernel<<<dim3(blocks), dim3(256), 0, stream>>>(x, h, src, dst, sum_x, sum_h, deg, E);
    }

    // 4. build bf16 A matrix (divide by degree fused)
    {
        long long tot = (long long)N * 128;
        build_a_kernel<<<dim3((unsigned)((tot + 255) / 256)), dim3(256), 0, stream>>>(
            x, h, sum_x, sum_h, deg, Abf, N);
    }

    // 5. WMMA GEMM per gate, weights staged in 128 KB LDS
    {
        int mTiles = (N + 15) / 16;                 // 3125 (exact)
        dim3 grid((mTiles + 3) / 4, 4);             // 4 M-tiles per block, 2 col-halves
        gemm_kernel<<<grid, dim3(256), B_LDS_BYTES, stream>>>(Abf, Wcat, S, N);
    }

    // 6. fused gates -> (new_h, new_c)
    {
        long long tot = (long long)N * OUT_FEATS;
        lstm_kernel<<<dim3((unsigned)((tot + 255) / 256)), dim3(256), 0, stream>>>(
            S, c, w_ci, w_cf, w_co, b_i, b_f, b_c, b_o, out, N);
    }
}
